// StrategyAwareScalarScalarOps_44341242364554
// MI455X (gfx1250) — compile-verified, matrix operand order checked by reading, not run on hardware
//
#include <hip/hip_runtime.h>
#include <hip/hip_bf16.h>
#include <math.h>

// ---------------- problem constants ----------------
#define D        1024
#define NSTRAT   20
#define BATCH    4096
#define TOPK     4
#define M_ROWS   (BATCH * TOPK)      // 16384

typedef unsigned short u16;
typedef __attribute__((ext_vector_type(16))) __bf16 v16bf;
typedef __attribute__((ext_vector_type(8)))  float  v8f;
typedef int v4i_vs __attribute__((vector_size(16)));   // matches builtin param type

// gfx1250 async global->LDS path (ASYNCcnt-tracked DMA) — probe-confirmed.
#if __has_builtin(__builtin_amdgcn_global_load_async_to_lds_b128) && \
    __has_builtin(__builtin_amdgcn_s_wait_asynccnt)
#define USE_ASYNC 1
#else
#define USE_ASYNC 0
#endif

#define AS1 __attribute__((address_space(1)))
#define AS3 __attribute__((address_space(3)))

// ---------------- helpers ----------------
__device__ __forceinline__ u16 f2bf(float f) {
  unsigned u = __float_as_uint(f);
  unsigned r = u + 0x7FFFu + ((u >> 16) & 1u);   // round-to-nearest-even
  if ((u & 0x7F800000u) == 0x7F800000u) r = u;   // inf/nan passthrough
  return (u16)(r >> 16);
}

__device__ __forceinline__ float safe_div(float num, float den) {
  float s = (den >= 0.0f) ? 1.0f : -1.0f;       // sign(0) -> +1 per reference
  return num / (s * (fabsf(den) + 0.1f));
}

#if USE_ASYNC
__device__ __forceinline__ void async_cp16(const u16* g, u16* l) {
  __builtin_amdgcn_global_load_async_to_lds_b128(
      (AS1 v4i_vs*)(uintptr_t)g,
      (AS3 v4i_vs*)(unsigned)(uintptr_t)l, 0, 0);
}
#endif

// ---------------- kernel 1: top-k + softmax weights ----------------
__global__ void topk_kernel(const float* __restrict__ scores,
                            const float* __restrict__ temp,
                            float* __restrict__ hdr_w,
                            int*   __restrict__ hdr_i,
                            int*   __restrict__ hdr_j) {
  if (threadIdx.x != 0 || blockIdx.x != 0) return;
  float vals[TOPK];
  int   idx[TOPK];
#pragma unroll
  for (int k = 0; k < TOPK; ++k) { vals[k] = -INFINITY; idx[k] = 0; }
  for (int i = 0; i < NSTRAT * NSTRAT; ++i) {
    float v = scores[i];
    if (v > vals[TOPK - 1]) {               // strict > keeps earliest index on ties
      int p = TOPK - 1;
      while (p > 0 && v > vals[p - 1]) {
        vals[p] = vals[p - 1]; idx[p] = idx[p - 1]; --p;
      }
      vals[p] = v; idx[p] = i;
    }
  }
  float tc = fmaxf(temp[0], 0.1f);
  float mx = vals[0];
  float e[TOPK], s = 0.0f;
#pragma unroll
  for (int k = 0; k < TOPK; ++k) { e[k] = __expf((vals[k] - mx) / tc); s += e[k]; }
#pragma unroll
  for (int k = 0; k < TOPK; ++k) {
    hdr_w[k] = e[k] / s;
    hdr_i[k] = idx[k] / NSTRAT;
    hdr_j[k] = idx[k] % NSTRAT;
  }
}

// ---------------- kernel 2: fp32 weight -> transposed bf16 ----------------
// in:  W[kdim][n] (row-major fp32), n = 1<<lgn.   out: Wt[n][kdim] (bf16)
__global__ void cvt_t_kernel(const float* __restrict__ W, u16* __restrict__ Wt,
                             int kdim, int lgn) {
  const size_t total  = (size_t)kdim << lgn;
  const unsigned mask = (1u << lgn) - 1u;
  const size_t stride = (size_t)gridDim.x * blockDim.x;
  for (size_t i = (size_t)blockIdx.x * blockDim.x + threadIdx.x; i < total; i += stride) {
    int k = (int)(i >> lgn);
    int c = (int)(i & mask);
    Wt[(size_t)c * kdim + k] = f2bf(W[i]);
  }
}

// ---------------- kernel 3: feature construction (bf16) ----------------
// mode 0: ratio [rab|rba|log_ratio] (3D)   mode 1: rel [d,|d|,sign] (3D)
// mode 2: corr [p|-p] (2D) + product into fus col-block 4D   mode 3: quad (3D)
__global__ void feat_kernel(const float* __restrict__ enc_a,
                            const float* __restrict__ enc_b,
                            const int* __restrict__ hdr_i,
                            const int* __restrict__ hdr_j,
                            u16* __restrict__ F, int fld,
                            u16* __restrict__ fus, int mode) {
  const size_t total  = (size_t)M_ROWS * D;
  const size_t stride = (size_t)gridDim.x * blockDim.x;
  for (size_t n = (size_t)blockIdx.x * blockDim.x + threadIdx.x; n < total; n += stride) {
    int d = (int)(n & (D - 1));
    int m = (int)(n >> 10);
    int b = m >> 2, k = m & 3;
    int ia = hdr_i[k], jb = hdr_j[k];
    float ea = enc_a[((size_t)b * NSTRAT + ia) * D + d];
    float eb = enc_b[((size_t)b * NSTRAT + jb) * D + d];
    size_t base = (size_t)m * fld + d;
    if (mode == 0) {
      float rab = safe_div(ea, eb);
      float rba = safe_div(eb, ea);
      float lr  = copysignf(log1pf(fabsf(rab)), rab);
      if (rab == 0.0f) lr = 0.0f;
      F[base]         = f2bf(rab);
      F[base + D]     = f2bf(rba);
      F[base + 2 * D] = f2bf(lr);
    } else if (mode == 1) {
      float df = ea - eb;
      float sg = (df > 0.0f) ? 1.0f : ((df < 0.0f) ? -1.0f : 0.0f);
      F[base]         = f2bf(df);
      F[base + D]     = f2bf(fabsf(df));
      F[base + 2 * D] = f2bf(sg);
    } else if (mode == 2) {
      float p = ea * eb;
      F[base]         = f2bf(p);
      F[base + D]     = f2bf(-p);
      fus[(size_t)m * (5 * D) + 4 * D + d] = f2bf(p);
    } else {
      float p = ea * eb;
      F[base]         = f2bf(ea * ea);
      F[base + D]     = f2bf(eb * eb);
      F[base + 2 * D] = f2bf(p);
    }
  }
}

// ---------------- kernel 4: bf16 WMMA GEMM ----------------
// C = act(A[M x kdim] @ W[kdim x N] + bias), W given transposed Wt[N][kdim].
// Block = 128 threads = 4 waves. Tile 128(M) x 64(N); LDS stage depth 64 (two
// k=32 sub-steps per buffer) -> 16 WMMAs per barrier pair.
#define TM 128
#define TN 64
#define TK 64
#define LDSS 72   // padded ushort stride (144B/row): rows spread over bank groups

__global__ __launch_bounds__(128)
void gemm_bf16_wmma(const u16* __restrict__ A, int lda,
                    const u16* __restrict__ Wt, int kdim,
                    const float* __restrict__ bias,
                    void* __restrict__ Cout, int ldc, int coloff,
                    int relu, int out_f32) {
  const int t    = threadIdx.x;
  const int wave = t >> 5;
  const int lane = t & 31;
  const int lr   = lane & 15;      // row (A) / col (B) within 16
  const int hi   = lane >> 4;      // K-half select per ISA 16-bit layout
  const int bm   = blockIdx.y * TM;
  const int bn   = blockIdx.x * TN;

  v8f acc0[4] = {};
  v8f acc1[4] = {};

  union Frag { v16bf v; struct { uint4 lo, hi; } q; };

  // A frag: lanes 0-15 hold K {0..7,16..23}; lanes 16-31 hold K {8..15,24..31}.
  // Preload all 4 B frags, then issue 8 back-to-back WMMAs (one DS wait).
  auto compute = [&](const u16* bufA, const u16* bufB, int ks) {
    Frag a0, a1, b0, b1, b2, b3;
    const u16* ap0 = &bufA[(wave * 32 + lr) * LDSS + ks + hi * 8];
    const u16* ap1 = ap0 + 16 * LDSS;
    a0.q.lo = *(const uint4*)(ap0);  a0.q.hi = *(const uint4*)(ap0 + 16);
    a1.q.lo = *(const uint4*)(ap1);  a1.q.hi = *(const uint4*)(ap1 + 16);
    const u16* bp = &bufB[lr * LDSS + ks + hi * 8];
    b0.q.lo = *(const uint4*)(bp);               b0.q.hi = *(const uint4*)(bp + 16);
    b1.q.lo = *(const uint4*)(bp + 16 * LDSS);   b1.q.hi = *(const uint4*)(bp + 16 * LDSS + 16);
    b2.q.lo = *(const uint4*)(bp + 32 * LDSS);   b2.q.hi = *(const uint4*)(bp + 32 * LDSS + 16);
    b3.q.lo = *(const uint4*)(bp + 48 * LDSS);   b3.q.hi = *(const uint4*)(bp + 48 * LDSS + 16);
    acc0[0] = __builtin_amdgcn_wmma_f32_16x16x32_bf16(false, a0.v, false, b0.v, (short)0, acc0[0], false, false);
    acc1[0] = __builtin_amdgcn_wmma_f32_16x16x32_bf16(false, a1.v, false, b0.v, (short)0, acc1[0], false, false);
    acc0[1] = __builtin_amdgcn_wmma_f32_16x16x32_bf16(false, a0.v, false, b1.v, (short)0, acc0[1], false, false);
    acc1[1] = __builtin_amdgcn_wmma_f32_16x16x32_bf16(false, a1.v, false, b1.v, (short)0, acc1[1], false, false);
    acc0[2] = __builtin_amdgcn_wmma_f32_16x16x32_bf16(false, a0.v, false, b2.v, (short)0, acc0[2], false, false);
    acc1[2] = __builtin_amdgcn_wmma_f32_16x16x32_bf16(false, a1.v, false, b2.v, (short)0, acc1[2], false, false);
    acc0[3] = __builtin_amdgcn_wmma_f32_16x16x32_bf16(false, a0.v, false, b3.v, (short)0, acc0[3], false, false);
    acc1[3] = __builtin_amdgcn_wmma_f32_16x16x32_bf16(false, a1.v, false, b3.v, (short)0, acc1[3], false, false);
  };

#if USE_ASYNC
  // ---- double-buffered async-DMA pipeline (ASYNCcnt) ----
  __shared__ u16 lA[2][TM * LDSS];
  __shared__ u16 lB[2][TN * LDSS];

  auto load_tile = [&](int buf, int k0) {
    // A: 128 rows x 8 segs = 1024 chunks of 16B; B: 64 x 8 = 512 chunks.
#pragma unroll
    for (int c = 0; c < 8; ++c) {
      int idx = t + c * 128, row = idx >> 3, seg = idx & 7;
      async_cp16(A + (size_t)(bm + row) * lda + k0 + seg * 8,
                 &lA[buf][row * LDSS + seg * 8]);
    }
#pragma unroll
    for (int c = 0; c < 4; ++c) {
      int idx = t + c * 128, row = idx >> 3, seg = idx & 7;
      async_cp16(Wt + (size_t)(bn + row) * kdim + k0 + seg * 8,
                 &lB[buf][row * LDSS + seg * 8]);
    }
  };

  const int nsteps = kdim / TK;
  load_tile(0, 0);
  for (int s = 0; s < nsteps; ++s) {
    const int cur = s & 1;
    if (s + 1 < nsteps) {
      load_tile(cur ^ 1, (s + 1) * TK);       // overlap next tile with this compute
      __builtin_amdgcn_s_wait_asynccnt(12);   // own 12 ops for `cur` complete (in-order)
    } else {
      __builtin_amdgcn_s_wait_asynccnt(0);
    }
    __syncthreads();
    compute(lA[cur], lB[cur], 0);
    compute(lA[cur], lB[cur], 32);
    __syncthreads();                          // all reads of `cur` done before reuse
  }
#else
  // ---- fallback: batched global->reg->LDS, single buffer ----
  __shared__ u16 lA1[TM * LDSS];
  __shared__ u16 lB1[TN * LDSS];
  for (int k0 = 0; k0 < kdim; k0 += TK) {
    uint4 ra[8], rb[4];
#pragma unroll
    for (int c = 0; c < 8; ++c) {
      int idx = t + c * 128, row = idx >> 3, seg = idx & 7;
      ra[c] = *(const uint4*)(A + (size_t)(bm + row) * lda + k0 + seg * 8);
    }
#pragma unroll
    for (int c = 0; c < 4; ++c) {
      int idx = t + c * 128, row = idx >> 3, seg = idx & 7;
      rb[c] = *(const uint4*)(Wt + (size_t)(bn + row) * kdim + k0 + seg * 8);
    }
    __syncthreads();
#pragma unroll
    for (int c = 0; c < 8; ++c) {
      int idx = t + c * 128, row = idx >> 3, seg = idx & 7;
      *(uint4*)(&lA1[row * LDSS + seg * 8]) = ra[c];
    }
#pragma unroll
    for (int c = 0; c < 4; ++c) {
      int idx = t + c * 128, row = idx >> 3, seg = idx & 7;
      *(uint4*)(&lB1[row * LDSS + seg * 8]) = rb[c];
    }
    if (k0 + TK < kdim) {
      __builtin_prefetch(A  + (size_t)(bm + (t >> 1)) * lda  + k0 + TK + (t & 1) * 32, 0, 3);
      __builtin_prefetch(Wt + (size_t)(bn + (t >> 1)) * kdim + k0 + TK + (t & 1) * 32, 0, 3);
    }
    __syncthreads();
    compute(lA1, lB1, 0);
    compute(lA1, lB1, 32);
    __syncthreads();
  }
#endif

  // epilogue: acc VGPR r -> row = r + hi*8, col = lr (per ISA C/D layout)
#pragma unroll
  for (int mh = 0; mh < 2; ++mh) {
#pragma unroll
    for (int nt = 0; nt < 4; ++nt) {
#pragma unroll
      for (int r = 0; r < 8; ++r) {
        int row = bm + wave * 32 + mh * 16 + hi * 8 + r;
        int col = bn + nt * 16 + lr;
        float v = (mh ? acc1[nt][r] : acc0[nt][r]) + bias[col];
        if (relu) v = fmaxf(v, 0.0f);
        size_t o = (size_t)row * ldc + coloff + col;
        if (out_f32) ((float*)Cout)[o] = v;
        else         ((u16*)Cout)[o]   = f2bf(v);
      }
    }
  }
}

// ---------------- kernel 5: layernorm + softmax-weighted blend ----------------
__global__ __launch_bounds__(128)
void finalize_kernel(const float* __restrict__ X,   // [M_ROWS, D]
                     const float* __restrict__ w,   // [4]
                     const float* __restrict__ g,
                     const float* __restrict__ bln,
                     float* __restrict__ out) {     // [BATCH, D]
  __shared__ float red[128];
  const int b = blockIdx.x;
  const int t = threadIdx.x;
  float accv[8];
#pragma unroll
  for (int i = 0; i < 8; ++i) accv[i] = 0.0f;

  for (int k = 0; k < TOPK; ++k) {
    const float* row = X + ((size_t)b * TOPK + k) * D;
    float x[8], s = 0.0f;
#pragma unroll
    for (int i = 0; i < 8; ++i) { x[i] = row[t + i * 128]; s += x[i]; }
    red[t] = s; __syncthreads();
    for (int st = 64; st > 0; st >>= 1) { if (t < st) red[t] += red[t + st]; __syncthreads(); }
    float mu = red[0] * (1.0f / D); __syncthreads();
    float vs = 0.0f;
#pragma unroll
    for (int i = 0; i < 8; ++i) { float dd = x[i] - mu; vs += dd * dd; }
    red[t] = vs; __syncthreads();
    for (int st = 64; st > 0; st >>= 1) { if (t < st) red[t] += red[t + st]; __syncthreads(); }
    float rsig = rsqrtf(red[0] * (1.0f / D) + 1e-5f); __syncthreads();
    float wk = w[k];
#pragma unroll
    for (int i = 0; i < 8; ++i) {
      int d = t + i * 128;
      accv[i] += wk * ((x[i] - mu) * rsig * g[d] + bln[d]);
    }
  }
#pragma unroll
  for (int i = 0; i < 8; ++i) out[(size_t)b * D + t + i * 128] = accv[i];
}

// ---------------- host orchestration ----------------
extern "C" void kernel_launch(void* const* d_in, const int* in_sizes, int n_in,
                              void* d_out, int out_size, void* d_ws, size_t ws_size,
                              hipStream_t stream) {
  (void)in_sizes; (void)n_in; (void)out_size; (void)ws_size;

  const float* enc_a  = (const float*)d_in[0];
  const float* enc_b  = (const float*)d_in[1];
  const float* scores = (const float*)d_in[2];
  const float* temp   = (const float*)d_in[3];
  const float* W1[5]  = { (const float*)d_in[4],  (const float*)d_in[8],
                          (const float*)d_in[12], (const float*)d_in[16],
                          (const float*)d_in[20] };
  const float* B1[5]  = { (const float*)d_in[5],  (const float*)d_in[9],
                          (const float*)d_in[13], (const float*)d_in[17],
                          (const float*)d_in[21] };
  const float* W2[5]  = { (const float*)d_in[6],  (const float*)d_in[10],
                          (const float*)d_in[14], (const float*)d_in[18],
                          (const float*)d_in[22] };
  const float* B2[5]  = { (const float*)d_in[7],  (const float*)d_in[11],
                          (const float*)d_in[15], (const float*)d_in[19],
                          (const float*)d_in[23] };
  const float* ln_g = (const float*)d_in[24];
  const float* ln_b = (const float*)d_in[25];

  // workspace carve-up (all offsets 256B-aligned)
  char* ws = (char*)d_ws;
  float* hdr_w = (float*)ws;
  int*   hdr_i = (int*)(ws + 16);
  int*   hdr_j = (int*)(ws + 32);
  size_t off = 256;

  const int k1[5]  = { 3 * D, 3 * D, 2 * D, 3 * D, 5 * D };  // W1 kdims
  const int n1[5]  = { D, D, D, D, 2 * D };                  // W1 out dims
  const int ln1[5] = { 10, 10, 10, 10, 11 };                 // log2(n1)
  const int k2[5]  = { D, D, D, D, 2 * D };                  // W2 kdims
  const int ln2[5] = { 10, 10, 10, 10, 10 };
  const int n2[5]  = { D, D, D, D, D };
  u16* W1t[5]; u16* W2t[5];
  for (int m = 0; m < 5; ++m) {
    W1t[m] = (u16*)(ws + off); off += (size_t)k1[m] * n1[m] * sizeof(u16);
    W2t[m] = (u16*)(ws + off); off += (size_t)k2[m] * n2[m] * sizeof(u16);
  }
  u16* Fbuf  = (u16*)(ws + off); off += (size_t)M_ROWS * 3 * D * sizeof(u16);
  u16* Hbuf  = (u16*)(ws + off); off += (size_t)M_ROWS * 2 * D * sizeof(u16);
  u16* FUS   = (u16*)(ws + off); off += (size_t)M_ROWS * 5 * D * sizeof(u16);
  float* O32 = (float*)(ws + off); off += (size_t)M_ROWS * D * sizeof(float);

  // 1) top-k + softmax weights
  topk_kernel<<<1, 32, 0, stream>>>(scores, temp, hdr_w, hdr_i, hdr_j);

  // 2) weight conversion + transpose (fp32 -> bf16, [k][n] -> [n][k])
  for (int m = 0; m < 5; ++m) {
    cvt_t_kernel<<<4096, 256, 0, stream>>>(W1[m], W1t[m], k1[m], ln1[m]);
    cvt_t_kernel<<<4096, 256, 0, stream>>>(W2[m], W2t[m], k2[m], ln2[m]);
  }

  // 3) per-MLP: features -> GEMM(relu) -> GEMM into fusion-input column block
  dim3 blk(128);
  for (int m = 0; m < 4; ++m) {          // 0 ratio, 1 rel, 2 corr, 3 quad
    int fld = (m == 2) ? 2 * D : 3 * D;
    feat_kernel<<<4096, 256, 0, stream>>>(enc_a, enc_b, hdr_i, hdr_j,
                                          Fbuf, fld, FUS, m);
    dim3 g1(D / TN, M_ROWS / TM);
    gemm_bf16_wmma<<<g1, blk, 0, stream>>>(Fbuf, fld, W1t[m], k1[m], B1[m],
                                           Hbuf, D, 0, /*relu=*/1, /*f32=*/0);
    gemm_bf16_wmma<<<g1, blk, 0, stream>>>(Hbuf, D, W2t[m], k2[m], B2[m],
                                           FUS, 5 * D, m * D, /*relu=*/0, /*f32=*/0);
  }

  // 4) fusion MLP: [M,5D] -> relu -> [M,2D] -> [M,D] (f32)
  {
    dim3 gA(2 * D / TN, M_ROWS / TM);
    gemm_bf16_wmma<<<gA, blk, 0, stream>>>(FUS, 5 * D, W1t[4], 5 * D, B1[4],
                                           Hbuf, 2 * D, 0, /*relu=*/1, /*f32=*/0);
    dim3 gB(D / TN, M_ROWS / TM);
    gemm_bf16_wmma<<<gB, blk, 0, stream>>>(Hbuf, 2 * D, W2t[4], 2 * D, B2[4],
                                           O32, D, 0, /*relu=*/0, /*f32=*/1);
  }

  // 5) layernorm + weighted blend over K -> [B, D]
  finalize_kernel<<<BATCH, 128, 0, stream>>>(O32, hdr_w, ln_g, ln_b, (float*)d_out);
}
